// MultiHeadSelfAttention_13786845020937
// MI455X (gfx1250) — matrix operand order repaired and checked by reading, not compiled, not run on hardware
//
#include <hip/hip_runtime.h>

typedef __attribute__((ext_vector_type(16))) _Float16 v16h;
typedef __attribute__((ext_vector_type(8)))  _Float16 v8h;
typedef __attribute__((ext_vector_type(8)))  float    v8f;
typedef __attribute__((ext_vector_type(4)))  int      v4i;
typedef __attribute__((address_space(1))) v4i gv4i;   // global
typedef __attribute__((address_space(3))) v4i lv4i;   // LDS

#define E_DIM 1024
#define NHEAD 16
#define DHEAD 64
#define BATCH 4
#define SEQ   2048
#define MROWS (BATCH * SEQ)   // 8192

// ---------------------------------------------------------------------------
// Async global->LDS copy (CDNA5 GLOBAL_LOAD_ASYNC_TO_LDS_B128, ASYNCcnt).
// Builtin prototype (from hipcc diagnostic): (v4i AS1*, v4i AS3*, Ii, Ii).
// Falls back to sync load + ds_store if the builtin is absent.
// ---------------------------------------------------------------------------
#if defined(__has_builtin)
#if __has_builtin(__builtin_amdgcn_global_load_async_to_lds_b128)
#define ASYNC_LDS 1
#endif
#endif

__device__ __forceinline__ void copy16_to_lds(const _Float16* g, _Float16* l) {
#ifdef ASYNC_LDS
    __builtin_amdgcn_global_load_async_to_lds_b128(
        (gv4i*)(void*)g, (lv4i*)(void*)l, 0, 0);
#else
    *(v8h*)l = *(const v8h*)g;
#endif
}

__device__ __forceinline__ void async_wait_all() {
#ifdef ASYNC_LDS
#if __has_builtin(__builtin_amdgcn_s_wait_asynccnt)
    __builtin_amdgcn_s_wait_asynccnt(0);
#else
    asm volatile("s_wait_asynccnt 0x0" ::: "memory");
#endif
#endif
}

// ---------------------------------------------------------------------------
// Fragment layouts per CDNA5 WMMA VGPR tables (CDNA5 ISA 7.12.2)
// A (16x32 f16): lane m = lane%16; elems 0..7  -> K = kb + hi*8 + e
//                                elems 8..15 -> K = kb + 16 + hi*8 + (e-8)
// B (32x16 f16): lane n = lane%16; same K order as A (hi*8 + e, 16 + hi*8 + (e-8))
// C (16x16 f32): lane half hi=lane/16; VGPR r -> (M = r + 8*hi, N = lane%16)
// ---------------------------------------------------------------------------
__device__ __forceinline__ v16h lds_a_frag(const _Float16* tile, int row,
                                           int nl, int hi) {
    const _Float16* p = tile + (row + nl) * 32 + hi * 8;
    v8h lo = *(const v8h*)p;
    v8h h2 = *(const v8h*)(p + 16);
    return __builtin_shufflevector(lo, h2, 0, 1, 2, 3, 4, 5, 6, 7,
                                   8, 9, 10, 11, 12, 13, 14, 15);
}

__device__ __forceinline__ v16h lds_b_frag(const _Float16* tile, int col,
                                           int nl, int hi) {
    const _Float16* p = tile + (col + nl) * 32 + hi * 8;
    v8h lo = *(const v8h*)p;
    v8h h2 = *(const v8h*)(p + 16);
    return __builtin_shufflevector(lo, h2, 0, 1, 2, 3, 4, 5, 6, 7,
                                   8, 9, 10, 11, 12, 13, 14, 15);
}

// ---------------------------------------------------------------------------
// X (f32) -> f16
// ---------------------------------------------------------------------------
__global__ __launch_bounds__(256) void cvt_f32_to_f16(const float* __restrict__ X,
                                                      _Float16* __restrict__ Xh) {
    const size_t i = (size_t)blockIdx.x * blockDim.x + threadIdx.x;
    float4 v = ((const float4*)X)[i];
    _Float16* o = Xh + i * 4;
    o[0] = (_Float16)v.x; o[1] = (_Float16)v.y;
    o[2] = (_Float16)v.z; o[3] = (_Float16)v.w;
}

// ---------------------------------------------------------------------------
// W [K][N] f32 -> WT [N][K] f16 (32x32 LDS tiles, coalesced both sides)
// ---------------------------------------------------------------------------
__global__ __launch_bounds__(256) void transpose_cvt(const float* __restrict__ W,
                                                     _Float16* __restrict__ WT) {
    __shared__ float tile[32][33];
    const int tx = threadIdx.x, ty = threadIdx.y;
    const int k0 = blockIdx.y * 32, n0 = blockIdx.x * 32;
#pragma unroll
    for (int i = 0; i < 4; ++i)
        tile[ty + i * 8][tx] = W[(size_t)(k0 + ty + i * 8) * E_DIM + n0 + tx];
    __syncthreads();
#pragma unroll
    for (int i = 0; i < 4; ++i)
        WT[(size_t)(n0 + ty + i * 8) * E_DIM + k0 + tx] =
            (_Float16)tile[tx][ty + i * 8];
}

// ---------------------------------------------------------------------------
// GEMM: C[M,N] = A[M,K](f16) * BT[N,K](f16) + bias
// mode 0: store f16 split-heads [B,H,S,D]   mode 1: store f32 [M,N]
// 256 thr = 8 waves; block tile 128x128; wave tile 32x64; K-step 32.
// A/B tiles double-buffered in LDS via async global->LDS copies.
// ---------------------------------------------------------------------------
__global__ __launch_bounds__(256)
void gemm_wmma(const _Float16* __restrict__ A, const _Float16* __restrict__ BT,
               const float* __restrict__ bias, _Float16* __restrict__ outh,
               float* __restrict__ outf, int mode) {
    __shared__ _Float16 sA[2][128 * 32];   // [row][k] per buffer (8 KB each)
    __shared__ _Float16 sB[2][128 * 32];   // [col][k] per buffer

    const int tid = threadIdx.x;
    const int lane = tid & 31;
    const int wave = tid >> 5;
    const int wm = wave & 3, wn = wave >> 2;
    const int rowB = blockIdx.x * 128;
    const int colB = blockIdx.y * 128;
    const int nl = lane & 15, hi = lane >> 4;

    // Stage one 128x32 K-slice of A and BT into LDS buffer `buf`.
    auto stage = [&](int buf, int kb) {
#pragma unroll
        for (int i = 0; i < 2; ++i) {
            const int c = tid * 2 + i;            // 0..511
            const int r = c >> 2;                 // 0..127
            const int kc = (c & 3) * 8;           // 0,8,16,24
            copy16_to_lds(A + (size_t)(rowB + r) * E_DIM + kb + kc,
                          &sA[buf][r * 32 + kc]);
            copy16_to_lds(BT + (size_t)(colB + r) * E_DIM + kb + kc,
                          &sB[buf][r * 32 + kc]);
        }
    };

    v8f acc[2][4];
#pragma unroll
    for (int mt = 0; mt < 2; ++mt)
#pragma unroll
        for (int nt = 0; nt < 4; ++nt) acc[mt][nt] = (v8f)0.0f;

    stage(0, 0);
    async_wait_all();
    __syncthreads();

    const int NK = E_DIM / 32;
    for (int ks = 0; ks < NK; ++ks) {
        const int buf = ks & 1;
        if (ks + 1 < NK) stage(buf ^ 1, (ks + 1) * 32);  // overlap with compute

        const _Float16* Abuf = sA[buf];
        const _Float16* Bbuf = sB[buf];
        v16h a0 = lds_a_frag(Abuf, wm * 32, nl, hi);
        v16h a1 = lds_a_frag(Abuf, wm * 32 + 16, nl, hi);
#pragma unroll
        for (int nt = 0; nt < 4; ++nt) {
            v16h bf = lds_b_frag(Bbuf, wn * 64 + nt * 16, nl, hi);
            acc[0][nt] = __builtin_amdgcn_wmma_f32_16x16x32_f16(
                false, a0, false, bf, (short)0, acc[0][nt], false, false);
            acc[1][nt] = __builtin_amdgcn_wmma_f32_16x16x32_f16(
                false, a1, false, bf, (short)0, acc[1][nt], false, false);
        }

        async_wait_all();   // next buffer resident
        __syncthreads();    // all waves done reading current buffer
    }

    const int row0 = rowB + wm * 32;
    const int col0 = colB + wn * 64;
#pragma unroll
    for (int mt = 0; mt < 2; ++mt)
#pragma unroll
        for (int nt = 0; nt < 4; ++nt) {
            const int col = col0 + nt * 16 + nl;
            const float bv = bias[col];
#pragma unroll
            for (int r = 0; r < 8; ++r) {
                const int row = row0 + mt * 16 + r + 8 * hi;
                const float v = acc[mt][nt][r] + bv;
                if (mode == 0) {
                    const int b = row >> 11, s = row & 2047;
                    const int h = col >> 6, d = col & 63;
                    outh[(((size_t)b * NHEAD + h) * SEQ + s) * DHEAD + d] =
                        (_Float16)v;
                } else {
                    outf[(size_t)row * E_DIM + col] = v;
                }
            }
        }
}

// ---------------------------------------------------------------------------
// Flash attention: one block = 64 query rows of one (b,h); 4 waves x 16 rows.
// Online softmax in exp2 domain (log2e folded into Q prescale).
// K tile async-staged to LDS (shared by all 4 waves); V^T scattered to LDS.
// ---------------------------------------------------------------------------
__global__ __launch_bounds__(128)
void flash_attn(const _Float16* __restrict__ Q, const _Float16* __restrict__ K,
                const _Float16* __restrict__ V, _Float16* __restrict__ Acc) {
    __shared__ _Float16 Pb[4 * 16 * 64];   // per-wave P tiles (C->A relayout)
    __shared__ _Float16 VT[64 * 64];       // V^T tile: [d][key]
    __shared__ _Float16 Ks[64 * 64];       // K tile:  [key][d]

    const int tid = threadIdx.x, lane = tid & 31, wave = tid >> 5;
    const int nl = lane & 15, hi = lane >> 4;
    const int bh = blockIdx.y;
    const int b = bh >> 4, h = bh & 15;
    const size_t base = (size_t)bh * SEQ * DHEAD;
    const _Float16* Qp = Q + base;
    const _Float16* Kp = K + base;
    const _Float16* Vp = V + base;
    const int q0 = blockIdx.x * 64 + wave * 16;
    _Float16* Pw = Pb + wave * 16 * 64;

    // Q fragments, prescaled by (1/sqrt(D)) * log2(e)
    const float qs = 0.125f * 1.44269504088896340736f;
    v16h aq[2];
#pragma unroll
    for (int t = 0; t < 2; ++t) {
        const _Float16* p = Qp + (size_t)(q0 + nl) * DHEAD + t * 32 + hi * 8;
        v8h lo = *(const v8h*)p;
        v8h h2 = *(const v8h*)(p + 16);
        v16h a = __builtin_shufflevector(lo, h2, 0, 1, 2, 3, 4, 5, 6, 7,
                                         8, 9, 10, 11, 12, 13, 14, 15);
#pragma unroll
        for (int i = 0; i < 16; ++i) a[i] = (_Float16)((float)a[i] * qs);
        aq[t] = a;
    }

    v8f o[4];
#pragma unroll
    for (int nt = 0; nt < 4; ++nt) o[nt] = (v8f)0.0f;
    float mr[8], lr[8];
#pragma unroll
    for (int r = 0; r < 8; ++r) { mr[r] = -3.0e38f; lr[r] = 0.0f; }

    for (int jb = 0; jb < SEQ; jb += 64) {
        __syncthreads();  // previous iteration done reading Ks / VT / P

        // Async-stage K tile (straight copy, layout already B-frag friendly)
#pragma unroll
        for (int i = 0; i < 4; ++i) {
            const int c = tid * 4 + i;            // 0..511 chunks of 8 halves
            copy16_to_lds(Kp + (size_t)jb * DHEAD + c * 8, Ks + c * 8);
        }
        // Stage V^T (transpose scatter through VGPRs)
#pragma unroll
        for (int i = 0; i < 4; ++i) {
            const int c = tid * 4 + i;
            const int key = c >> 3, dc = c & 7;
            v8h vv = *(const v8h*)(Vp + (size_t)(jb + key) * DHEAD + dc * 8);
#pragma unroll
            for (int e = 0; e < 8; ++e) VT[(dc * 8 + e) * 64 + key] = vv[e];
        }
        async_wait_all();
        __syncthreads();  // Ks + VT resident for all waves

        // Scores S = (Q * log2e/sqrt(D)) . K^T  (16 x 64)
        v8f sf[4];
#pragma unroll
        for (int nt = 0; nt < 4; ++nt) sf[nt] = (v8f)0.0f;
#pragma unroll
        for (int nt = 0; nt < 4; ++nt)
#pragma unroll
            for (int t = 0; t < 2; ++t) {
                const _Float16* kp = Ks + (nt * 16 + nl) * DHEAD + t * 32 + hi * 8;
                v16h bk = __builtin_shufflevector(*(const v8h*)kp, *(const v8h*)(kp + 16),
                                                  0, 1, 2, 3, 4, 5, 6, 7,
                                                  8, 9, 10, 11, 12, 13, 14, 15);
                sf[nt] = __builtin_amdgcn_wmma_f32_16x16x32_f16(
                    false, aq[t], false, bk, (short)0, sf[nt], false, false);
            }

        // Online softmax (rows live in 16-lane halves of each C VGPR)
        float mnew[8], al[8], rs[8];
#pragma unroll
        for (int r = 0; r < 8; ++r) {
            float v = fmaxf(fmaxf(sf[0][r], sf[1][r]), fmaxf(sf[2][r], sf[3][r]));
            v = fmaxf(v, __shfl_xor(v, 8));
            v = fmaxf(v, __shfl_xor(v, 4));
            v = fmaxf(v, __shfl_xor(v, 2));
            v = fmaxf(v, __shfl_xor(v, 1));
            mnew[r] = fmaxf(mr[r], v);
            al[r] = __builtin_exp2f(mr[r] - mnew[r]);
            mr[r] = mnew[r];
            rs[r] = 0.0f;
        }
#pragma unroll
        for (int nt = 0; nt < 4; ++nt)
#pragma unroll
            for (int r = 0; r < 8; ++r) {
                const float p = __builtin_exp2f(sf[nt][r] - mnew[r]);
                rs[r] += p;
                Pw[(r + 8 * hi) * 64 + nt * 16 + nl] = (_Float16)p;
            }
#pragma unroll
        for (int r = 0; r < 8; ++r) {
            float v = rs[r];
            v += __shfl_xor(v, 8);
            v += __shfl_xor(v, 4);
            v += __shfl_xor(v, 2);
            v += __shfl_xor(v, 1);
            lr[r] = lr[r] * al[r] + v;
        }
#pragma unroll
        for (int nt = 0; nt < 4; ++nt)
#pragma unroll
            for (int r = 0; r < 8; ++r) o[nt][r] *= al[r];

        // O += P . V  (A-fragments of P from own-wave LDS region, B from VT)
#pragma unroll
        for (int t = 0; t < 2; ++t) {
            const _Float16* pp = Pw + nl * 64 + t * 32 + hi * 8;
            v8h lo = *(const v8h*)pp;
            v8h h2 = *(const v8h*)(pp + 16);
            v16h ap = __builtin_shufflevector(lo, h2, 0, 1, 2, 3, 4, 5, 6, 7,
                                              8, 9, 10, 11, 12, 13, 14, 15);
#pragma unroll
            for (int nt = 0; nt < 4; ++nt) {
                const _Float16* vp = &VT[(nt * 16 + nl) * 64 + t * 32 + hi * 8];
                v16h bv2 = __builtin_shufflevector(*(const v8h*)vp, *(const v8h*)(vp + 16),
                                                   0, 1, 2, 3, 4, 5, 6, 7,
                                                   8, 9, 10, 11, 12, 13, 14, 15);
                o[nt] = __builtin_amdgcn_wmma_f32_16x16x32_f16(
                    false, ap, false, bv2, (short)0, o[nt], false, false);
            }
        }
    }

    // Normalize and store concat layout: Acc[b*S + s][h*64 + d] (f16)
#pragma unroll
    for (int nt = 0; nt < 4; ++nt)
#pragma unroll
        for (int r = 0; r < 8; ++r) {
            const int row = blockIdx.x * 64 + wave * 16 + r + 8 * hi;
            const int col = h * 64 + nt * 16 + nl;
            Acc[((size_t)b * SEQ + row) * E_DIM + col] =
                (_Float16)(o[nt][r] / lr[r]);
        }
}

// ---------------------------------------------------------------------------
extern "C" void kernel_launch(void* const* d_in, const int* in_sizes, int n_in,
                              void* d_out, int out_size, void* d_ws, size_t ws_size,
                              hipStream_t stream) {
    (void)in_sizes; (void)n_in; (void)out_size; (void)ws_size;
    const float* X  = (const float*)d_in[0];
    const float* Wq = (const float*)d_in[1];
    const float* bq = (const float*)d_in[2];
    const float* Wk = (const float*)d_in[3];
    const float* bk = (const float*)d_in[4];
    const float* Wv = (const float*)d_in[5];
    const float* bv = (const float*)d_in[6];
    const float* Wo = (const float*)d_in[7];
    const float* bo = (const float*)d_in[8];

    const size_t ME = (size_t)MROWS * E_DIM;  // 8M elems
    const size_t EE = (size_t)E_DIM * E_DIM;  // 1M elems
    _Float16* ws  = (_Float16*)d_ws;
    _Float16* Xh  = ws;            // reused as attn-concat buffer after QKV
    _Float16* WqT = ws + ME;
    _Float16* WkT = WqT + EE;
    _Float16* WvT = WkT + EE;
    _Float16* WoT = WvT + EE;
    _Float16* Qh  = WoT + EE;
    _Float16* Kh  = Qh + ME;
    _Float16* Vh  = Kh + ME;

    cvt_f32_to_f16<<<(int)(ME / 4 / 256), 256, 0, stream>>>(X, Xh);

    dim3 tB(32, 8), tG(32, 32);
    transpose_cvt<<<tG, tB, 0, stream>>>(Wq, WqT);
    transpose_cvt<<<tG, tB, 0, stream>>>(Wk, WkT);
    transpose_cvt<<<tG, tB, 0, stream>>>(Wv, WvT);
    transpose_cvt<<<tG, tB, 0, stream>>>(Wo, WoT);

    dim3 gG(MROWS / 128, E_DIM / 128);
    gemm_wmma<<<gG, 256, 0, stream>>>(Xh, WqT, bq, Qh, nullptr, 0);
    gemm_wmma<<<gG, 256, 0, stream>>>(Xh, WkT, bk, Kh, nullptr, 0);
    gemm_wmma<<<gG, 256, 0, stream>>>(Xh, WvT, bv, Vh, nullptr, 0);

    flash_attn<<<dim3(SEQ / 64, BATCH * NHEAD), 128, 0, stream>>>(Qh, Kh, Vh, Xh);

    gemm_wmma<<<gG, 256, 0, stream>>>(Xh, WoT, bo, nullptr, (float*)d_out, 1);
}
